// DAMHLoss_86784109183043
// MI455X (gfx1250) — matrix operand order repaired and checked by reading, not compiled, not run on hardware
//
#include <hip/hip_runtime.h>
#include <hip/hip_bf16.h>
#include <math.h>

// ---------------- problem constants (from reference) ----------------
#define N_PTS   6144
#define NBIT    128
#define NCLS    200
#define NT      (N_PTS / 16)          // 384 column tiles
#define WSTEPS  (NT / 4)              // 96 tiles per wave (4 waves/block)
#define UB_F    64.0f                 // BIT/2
#define C_COEF_F (-0.28719499f)       // (1/16)*ln(1/99)
#define A_COEF_F 2.0f                 // exact for y_p = 0.5
#define ALPHA_F  0.1f
#define NBINS   256                   // histogram bins over inner in [-128,128)

typedef __attribute__((ext_vector_type(16))) _Float16 v16h;
typedef __attribute__((ext_vector_type(8)))  _Float16 v8h;
typedef __attribute__((ext_vector_type(8)))  float    v8f;

// fast softplus: max(f,0) + ln2 * log2(1 + exp2(-|f|/ln2)); 2 hardware trans ops
__device__ __forceinline__ float softplus_f(float f) {
    float e = __builtin_amdgcn_exp2f(-1.4426950f * fabsf(f));
    return fmaxf(f, 0.0f) + 0.6931472f * __builtin_amdgcn_logf(1.0f + e);
}
__device__ __forceinline__ float clip_ub(float v) {
    return fminf(fmaxf(v, 0.0f), UB_F);
}
// v in [-128,128] -> v+128 >= 0, truncation == floor
__device__ __forceinline__ int bin_of(float v) {
    int b = (int)(v + 128.0f);
    return b < 0 ? 0 : (b > NBINS - 1 ? NBINS - 1 : b);
}

// A-operand fragment: 16-bit A 16x32 layout (lanes 0-15: M, K=0..7 & 16..23;
// lanes 16-31: K=8..15 & 24..31), shifted by 32*kk along K.
__device__ __forceinline__ v16h load_a_frag(const _Float16* __restrict__ row, int hi, int kk) {
    v16h a;
    ((v8h*)&a)[0] = *(const v8h*)(row +      8 * hi + 32 * kk);
    ((v8h*)&a)[1] = *(const v8h*)(row + 16 + 8 * hi + 32 * kk);
    return a;
}
// B-operand fragment: 16-bit B 32x16 layout (lanes 0-15 hold K=0..7 & 16..23,
// lanes 16-31 hold K=8..15 & 24..31), B[k][n] = uh[jBase+n][k] (B = uh^T).
__device__ __forceinline__ v16h load_b_frag(const _Float16* __restrict__ row, int hi, int kk) {
    v16h b;
    ((v8h*)&b)[0] = *(const v8h*)(row +      8 * hi + 32 * kk);
    ((v8h*)&b)[1] = *(const v8h*)(row + 16 + 8 * hi + 32 * kk);
    return b;
}

// ---------------- kernel 0: zero accumulators ----------------
__global__ void k_zero(int* __restrict__ ccount, float* __restrict__ acc) {
    int t = threadIdx.x;
    if (t < 256) ccount[t] = 0;
    if (t < 8)   acc[t] = 0.0f;
}

// ---------------- kernel 1: tanh + f16 pack + q-loss ----------------
__global__ void k_tanh(const float* __restrict__ u, _Float16* __restrict__ uh,
                       float* __restrict__ acc) {
    __shared__ float red[256];
    int tid = threadIdx.x;
    int idx = blockIdx.x * 256 + tid;
    float q = 0.0f;
    if (idx < N_PTS * NBIT) {
        float t = tanhf(u[idx]);
        uh[idx] = (_Float16)t;
        float sg = (t > 0.0f) ? 1.0f : ((t < 0.0f) ? -1.0f : 0.0f);
        float d = t - sg;
        q = d * d;
    }
    red[tid] = q;
    __syncthreads();
    for (int s = 128; s > 0; s >>= 1) {
        if (tid < s) red[tid] += red[tid + s];
        __syncthreads();
    }
    if (tid == 0) atomicAdd(&acc[0], red[0]);
}

// ---------------- kernel 2: labels from one-hot + class counts ----------------
__global__ void k_label(const int* __restrict__ y, int* __restrict__ labels,
                        int* __restrict__ ccount) {
    int i = blockIdx.x * 256 + threadIdx.x;
    if (i >= N_PTS) return;
    const int* row = y + (size_t)i * NCLS;
    int lab = 0;
    for (int c = 0; c < NCLS; ++c)
        if (row[c] != 0) lab = c;
    labels[i] = lab;
    atomicAdd(&ccount[lab], 1);
}

// ---------------- kernel 3 (pass A): WMMA GEMM + histograms -> BP/BPd ----------------
// 1 workgroup = 16-row strip, 4 waves sweep all 384 column tiles.
__global__ void __launch_bounds__(128)
k_stats(const _Float16* __restrict__ uh, const int* __restrict__ labels,
        const int* __restrict__ ccount, float* __restrict__ BP, float* __restrict__ BPd) {
    __shared__ int   histD[16 * NBINS];   // dissimilar value histogram per row
    __shared__ int   histS[16 * NBINS];   // similar value histogram per row
    __shared__ float redS[16], redD[16];
    __shared__ int   labRow[16];

    const int tid   = threadIdx.x;
    const int wave  = tid >> 5;
    const int lane  = tid & 31;
    const int hi    = lane >> 4;          // lane group (0/1)
    const int lo    = lane & 15;          // M (A) / N (B) within tile
    const int iBase = blockIdx.x * 16;

    for (int i = tid; i < 16 * NBINS; i += 128) { histD[i] = 0; histS[i] = 0; }
    if (tid < 16) {
        redS[tid] = 0.0f; redD[tid] = 0.0f;
        labRow[tid] = labels[iBase + tid];
    }
    __syncthreads();

    // per-lane register state: labels of the 8 rows this lane's outputs hit
    int labR[8];
    #pragma unroll
    for (int r = 0; r < 8; ++r) labR[r] = labRow[r + 8 * hi];
    float accS[8], accD[8];
    #pragma unroll
    for (int r = 0; r < 8; ++r) { accS[r] = 0.0f; accD[r] = 0.0f; }

    // A fragments for this strip (reused across every column tile)
    const _Float16* rowA = uh + (size_t)(iBase + lo) * NBIT;
    v16h a0 = load_a_frag(rowA, hi, 0);
    v16h a1 = load_a_frag(rowA, hi, 1);
    v16h a2 = load_a_frag(rowA, hi, 2);
    v16h a3 = load_a_frag(rowA, hi, 3);

    // software pipeline: preload B for first tile
    int jt = wave;
    const _Float16* rowB = uh + (size_t)(jt * 16 + lo) * NBIT;
    v16h b0 = load_b_frag(rowB, hi, 0);
    v16h b1 = load_b_frag(rowB, hi, 1);
    v16h b2 = load_b_frag(rowB, hi, 2);
    v16h b3 = load_b_frag(rowB, hi, 3);

    // exactly WSTEPS iterations per wave; all control flow is scalar-uniform,
    // EXEC stays all-ones (required by WMMA)
    for (int t = 0; t < WSTEPS; ++t) {
        const int labj = labels[jt * 16 + lo];

        v8f c = {};
        c = __builtin_amdgcn_wmma_f32_16x16x32_f16(false, a0, false, b0, (short)0, c, false, false);
        c = __builtin_amdgcn_wmma_f32_16x16x32_f16(false, a1, false, b1, (short)0, c, false, false);
        c = __builtin_amdgcn_wmma_f32_16x16x32_f16(false, a2, false, b2, (short)0, c, false, false);
        c = __builtin_amdgcn_wmma_f32_16x16x32_f16(false, a3, false, b3, (short)0, c, false, false);

        // branchless pipeline: last iteration re-loads the current tile
        const int jn = jt + 4;
        const int jp = (t + 1 < WSTEPS) ? jn : jt;   // scalar select, no EXEC change
        rowB = uh + (size_t)(jp * 16 + lo) * NBIT;
        __builtin_prefetch(rowB, 0, 3);
        b0 = load_b_frag(rowB, hi, 0);
        b1 = load_b_frag(rowB, hi, 1);
        b2 = load_b_frag(rowB, hi, 2);
        b3 = load_b_frag(rowB, hi, 3);

        #pragma unroll
        for (int r = 0; r < 8; ++r) {
            const int mi = r + 8 * hi;            // row within strip
            const float v = c[r];
            const int bin = bin_of(v);
            const bool sim = (labR[r] == labj);
            if (sim) accS[r] += v; else accD[r] += v;
            int* h = sim ? &histS[mi * NBINS + bin] : &histD[mi * NBINS + bin];
            atomicAdd(h, 1);
        }
        jt = jn;
    }

    // cross-lane / cross-wave reduction of the register sums (once)
    #pragma unroll
    for (int r = 0; r < 8; ++r) {
        const int mi = r + 8 * hi;
        atomicAdd(&redS[mi], accS[r]);
        atomicAdd(&redD[mi], accD[r]);
    }
    __syncthreads();

    if (tid < 16) {
        const int gi   = iBase + tid;
        const int nsim = ccount[labRow[tid]];
        const int ndis = N_PTS - nsim;
        const float meanS  = clip_ub(redS[tid] / fmaxf((float)nsim, 1.0f));
        const float meanDS = clip_ub(redD[tid] / fmaxf((float)ndis, 1.0f));

        // top ~10% of dissimilar inners (histogram tail, bin-center approx)
        const int tail = ndis - (ndis * 9) / 10;
        float tsum = 0.0f; int got = 0;
        for (int b = NBINS - 1; b >= 0 && got < tail; --b) {
            int cbin = histD[tid * NBINS + b];
            int take = cbin < (tail - got) ? cbin : (tail - got);
            tsum += (float)take * ((float)(b - 128) + 0.5f);
            got += take;
        }
        const float dMax = clip_ub(tsum / fmaxf((float)tail, 1.0f));

        // bottom ~10% of similar inners
        const int head = nsim - (nsim * 9) / 10;
        float hsum = 0.0f; got = 0;
        for (int b = 0; b < NBINS && got < head; ++b) {
            int cbin = histS[tid * NBINS + b];
            int take = cbin < (head - got) ? cbin : (head - got);
            hsum += (float)take * ((float)(b - 128) + 0.5f);
            got += take;
        }
        const float sMin = clip_ub(hsum / fmaxf((float)head, 1.0f));

        BP[gi]  = meanS  - (UB_F - meanS) / UB_F * fabsf(meanS - dMax);
        BPd[gi] = meanDS + meanDS / UB_F * fabsf(meanDS - sMin);
    }
}

// ---------------- kernel 4 (pass C): WMMA GEMM + masked softplus losses ----------------
__global__ void __launch_bounds__(128)
k_loss(const _Float16* __restrict__ uh, const int* __restrict__ labels,
       const int* __restrict__ ccount, const float* __restrict__ BP,
       const float* __restrict__ BPd, float* __restrict__ acc) {
    __shared__ float posR[16], negR[16];
    __shared__ int   csR[16], cdR[16];
    __shared__ int   labRow[16];
    __shared__ float bpRow[16], bpdRow[16];

    const int tid   = threadIdx.x;
    const int wave  = tid >> 5;
    const int lane  = tid & 31;
    const int hi    = lane >> 4;
    const int lo    = lane & 15;
    const int iBase = blockIdx.x * 16;

    if (tid < 16) {
        posR[tid] = 0.0f; negR[tid] = 0.0f; csR[tid] = 0; cdR[tid] = 0;
        labRow[tid] = labels[iBase + tid];
        bpRow[tid]  = BP[iBase + tid];
        bpdRow[tid] = BPd[iBase + tid];
    }
    __syncthreads();

    // per-lane register state for the 8 rows this lane touches
    int   labR[8]; float bpR[8], bpdR[8];
    #pragma unroll
    for (int r = 0; r < 8; ++r) {
        labR[r] = labRow[r + 8 * hi];
        bpR[r]  = bpRow[r + 8 * hi];
        bpdR[r] = bpdRow[r + 8 * hi];
    }
    float accP[8], accN[8]; int accCs[8], accCd[8];
    #pragma unroll
    for (int r = 0; r < 8; ++r) { accP[r] = 0.0f; accN[r] = 0.0f; accCs[r] = 0; accCd[r] = 0; }

    const _Float16* rowA = uh + (size_t)(iBase + lo) * NBIT;
    v16h a0 = load_a_frag(rowA, hi, 0);
    v16h a1 = load_a_frag(rowA, hi, 1);
    v16h a2 = load_a_frag(rowA, hi, 2);
    v16h a3 = load_a_frag(rowA, hi, 3);

    int jt = wave;
    const _Float16* rowB = uh + (size_t)(jt * 16 + lo) * NBIT;
    v16h b0 = load_b_frag(rowB, hi, 0);
    v16h b1 = load_b_frag(rowB, hi, 1);
    v16h b2 = load_b_frag(rowB, hi, 2);
    v16h b3 = load_b_frag(rowB, hi, 3);

    for (int t = 0; t < WSTEPS; ++t) {
        const int labj = labels[jt * 16 + lo];

        v8f c = {};
        c = __builtin_amdgcn_wmma_f32_16x16x32_f16(false, a0, false, b0, (short)0, c, false, false);
        c = __builtin_amdgcn_wmma_f32_16x16x32_f16(false, a1, false, b1, (short)0, c, false, false);
        c = __builtin_amdgcn_wmma_f32_16x16x32_f16(false, a2, false, b2, (short)0, c, false, false);
        c = __builtin_amdgcn_wmma_f32_16x16x32_f16(false, a3, false, b3, (short)0, c, false, false);

        const int jn = jt + 4;
        const int jp = (t + 1 < WSTEPS) ? jn : jt;   // branchless clamp
        rowB = uh + (size_t)(jp * 16 + lo) * NBIT;
        __builtin_prefetch(rowB, 0, 3);
        b0 = load_b_frag(rowB, hi, 0);
        b1 = load_b_frag(rowB, hi, 1);
        b2 = load_b_frag(rowB, hi, 2);
        b3 = load_b_frag(rowB, hi, 3);

        #pragma unroll
        for (int r = 0; r < 8; ++r) {
            const float v = c[r];
            if (labR[r] == labj) {
                const float bp = bpR[r];
                if (v != bp) {
                    const float dc = v - bp;
                    const float f  = (v > bp) ? C_COEF_F * dc : A_COEF_F * C_COEF_F * dc;
                    accP[r] += softplus_f(f);
                    accCs[r] += 1;
                }
            } else {
                const float bpd = bpdR[r];
                if (v != bpd) {
                    const float dc = v - bpd;
                    const float f  = (v < bpd) ? C_COEF_F * dc : A_COEF_F * C_COEF_F * dc;
                    accN[r] += softplus_f(-f);
                    accCd[r] += 1;
                }
            }
        }
        jt = jn;
    }

    // one-shot LDS reduction of register accumulators
    #pragma unroll
    for (int r = 0; r < 8; ++r) {
        const int mi = r + 8 * hi;
        atomicAdd(&posR[mi], accP[r]);
        atomicAdd(&negR[mi], accN[r]);
        atomicAdd(&csR[mi], accCs[r]);
        atomicAdd(&cdR[mi], accCd[r]);
    }
    __syncthreads();

    if (tid < 16) {
        const int nsim = ccount[labRow[tid]];
        const int ndis = N_PTS - nsim;
        const float rp = posR[tid] / fmaxf((float)csR[tid], 1.0f);
        const float rn = negR[tid] / fmaxf((float)cdR[tid], 1.0f);
        if (nsim > 0 && ndis > 0) {
            atomicAdd(&acc[1], rp);
            atomicAdd(&acc[2], rn);
            atomicAdd(&acc[3], 1.0f);
        }
    }
}

// ---------------- kernel 5: finalize scalar ----------------
__global__ void k_final(const float* __restrict__ acc, float* __restrict__ out) {
    const float cnt  = acc[3];
    const float posL = (cnt > 0.0f) ? acc[1] / cnt : 0.0f;
    const float navL = (cnt > 0.0f) ? acc[2] / cnt : 0.0f;
    const float q    = ALPHA_F * acc[0] / (float)(N_PTS * NBIT);
    out[0] = posL + navL + q;
}

// ---------------- host-side launch ----------------
extern "C" void kernel_launch(void* const* d_in, const int* in_sizes, int n_in,
                              void* d_out, int out_size, void* d_ws, size_t ws_size,
                              hipStream_t stream) {
    (void)in_sizes; (void)n_in; (void)out_size; (void)ws_size;
    const float* u = (const float*)d_in[0];   // [N, 128] f32
    const int*   y = (const int*)d_in[1];     // [N, 200] int32 one-hot
    // d_in[2] (ind) is unused by the reference.

    char* ws = (char*)d_ws;
    size_t off = 0;
    _Float16* uh     = (_Float16*)(ws + off); off += (size_t)N_PTS * NBIT * sizeof(_Float16); // 1.5 MB
    int*      labels = (int*)(ws + off);      off += (size_t)N_PTS * sizeof(int);
    int*      ccount = (int*)(ws + off);      off += 256 * sizeof(int);
    float*    BP     = (float*)(ws + off);    off += (size_t)N_PTS * sizeof(float);
    float*    BPd    = (float*)(ws + off);    off += (size_t)N_PTS * sizeof(float);
    float*    acc    = (float*)(ws + off);    off += 256;   // [0]=q, [1]=pos, [2]=nav, [3]=valid

    k_zero <<<1, 512, 0, stream>>>(ccount, acc);
    k_tanh <<<(N_PTS * NBIT + 255) / 256, 256, 0, stream>>>(u, uh, acc);
    k_label<<<(N_PTS + 255) / 256, 256, 0, stream>>>(y, labels, ccount);
    k_stats<<<N_PTS / 16, 128, 0, stream>>>(uh, labels, ccount, BP, BPd);
    k_loss <<<N_PTS / 16, 128, 0, stream>>>(uh, labels, ccount, BP, BPd, acc);
    k_final<<<1, 1, 0, stream>>>(acc, (float*)d_out);
}
